// WeldLSTM_10307921511128
// MI455X (gfx1250) — matrix operand order repaired and checked by reading, not compiled, not run on hardware
//
#include <hip/hip_runtime.h>
#include <math.h>

#define H 2048
#define T 4096
#define NWG 128
#define WGSIZE 256

typedef __attribute__((ext_vector_type(16))) __bf16 v16bf;
typedef __attribute__((ext_vector_type(8)))  float  v8f;

__device__ __forceinline__ unsigned short f2bf(float f) {
  unsigned u = __float_as_uint(f);
  u += 0x7FFFu + ((u >> 16) & 1u);          // round-to-nearest-even
  return (unsigned short)(u >> 16);
}

__device__ __forceinline__ float sigmoidf(float x) {
  return 1.0f / (1.0f + __expf(-x));
}

// ---------------------------------------------------------------------------
// Prologue: convert W_hh (8192x2048 f32, row-major) to bf16 pre-packed in the
// exact WMMA 16x32 A-tile lane layout, ordered so each wave streams its tiles
// contiguously (1KB per tile, lane L owns bytes [32L, 32L+32) of each tile).
// Tile id: blk = rb*64 + kt, rb = gate*128 + wg (16 rows each), kt = K/32.
// ---------------------------------------------------------------------------
__global__ void pack_whh_kernel(const float* __restrict__ Whh,
                                unsigned short* __restrict__ Wp) {
  unsigned e    = blockIdx.x * 256u + threadIdx.x;   // 0 .. 16777215
  unsigned blk  = e >> 9;                            // 512 bf16 per tile
  unsigned rem  = e & 511u;
  unsigned lane = rem >> 4;                          // 0..31
  unsigned idx  = rem & 15u;                         // bf16 index within lane
  unsigned half = lane >> 4;                         // 0: K 0-7/16-23, 1: K 8-15/24-31
  unsigned mrow = lane & 15u;
  unsigned chunk = idx >> 3;                         // which b128 of the lane
  unsigned koff  = idx & 7u;
  unsigned k_local = chunk * 16u + half * 8u + koff;
  unsigned rb = blk >> 6;                            // 0..511 (16-row block)
  unsigned kt = blk & 63u;
  unsigned q  = rb >> 7;                             // gate 0..3 (i,f,g,o)
  unsigned wg = rb & 127u;
  unsigned row = q * 2048u + wg * 16u + mrow;
  unsigned k   = kt * 32u + k_local;
  Wp[e] = f2bf(Whh[row * 2048u + k]);
}

// ---------------------------------------------------------------------------
// Device-wide barrier (sense-relative; counters zeroed by memset each call).
// ---------------------------------------------------------------------------
__device__ __forceinline__ void grid_sync(unsigned* bar) {
  __threadfence();
  __syncthreads();
  if (threadIdx.x == 0) {
    unsigned* cnt = bar;
    unsigned* gen = bar + 1;
    unsigned g0 = __hip_atomic_load(gen, __ATOMIC_RELAXED, __HIP_MEMORY_SCOPE_AGENT);
    unsigned prev = __hip_atomic_fetch_add(cnt, 1u, __ATOMIC_ACQ_REL, __HIP_MEMORY_SCOPE_AGENT);
    if (prev == (unsigned)(NWG - 1)) {
      __hip_atomic_store(cnt, 0u, __ATOMIC_RELAXED, __HIP_MEMORY_SCOPE_AGENT);
      __hip_atomic_fetch_add(gen, 1u, __ATOMIC_RELEASE, __HIP_MEMORY_SCOPE_AGENT);
    } else {
      while (__hip_atomic_load(gen, __ATOMIC_ACQUIRE, __HIP_MEMORY_SCOPE_AGENT) == g0) {
        __builtin_amdgcn_s_sleep(2);
      }
    }
  }
  __syncthreads();
  __threadfence();
}

// ---------------------------------------------------------------------------
// Persistent LSTM kernel: 128 WGs x 256 threads, one grid barrier per step.
// WG w owns h-elements [16w, 16w+16). Wave v: gate q=v&3, K-half kh=v>>2.
// ---------------------------------------------------------------------------
__global__ void __launch_bounds__(WGSIZE)
lstm_persistent(const float* __restrict__ src,
                const float* __restrict__ W_ih,
                const float* __restrict__ b_ih,
                const float* __restrict__ b_hh,
                const float* __restrict__ W_lin,
                const float* __restrict__ b_lin,
                const unsigned short* __restrict__ Wp,
                unsigned short* __restrict__ hb,     // 2 x 2048 bf16 (double buf)
                float* __restrict__ partials,        // 2 x 128 (double buf)
                unsigned* __restrict__ bar,
                float* __restrict__ out) {
  __shared__ __align__(16) unsigned short hb_lds[H];  // staged h (bf16)
  __shared__ float gbuf[2][64];                       // [K-half][gate*16 + m]

  const int tid  = threadIdx.x;
  const int lane = tid & 31;
  const int wv   = tid >> 5;      // 0..7
  const int q4   = wv & 3;        // gate
  const int kh   = wv >> 2;       // K half
  const int wg   = blockIdx.x;

  // Per-element constants for the updating lanes (wave 0, lanes 0..15).
  float bias_c[4] = {0.f, 0.f, 0.f, 0.f};
  float wxa[4]    = {0.f, 0.f, 0.f, 0.f};
  float wxb[4]    = {0.f, 0.f, 0.f, 0.f};
  float wxc[4]    = {0.f, 0.f, 0.f, 0.f};
  float wl = 0.f, cstate = 0.f;
  const float blin = b_lin[0];
  if (wv == 0 && lane < 16) {
    int j = wg * 16 + lane;
#pragma unroll
    for (int q = 0; q < 4; ++q) {
      int r = q * 2048 + j;
      bias_c[q] = b_ih[r] + b_hh[r];
      wxa[q] = W_ih[r * 3 + 0];
      wxb[q] = W_ih[r * 3 + 1];
      wxc[q] = W_ih[r * 3 + 2];
    }
    wl = W_lin[j];
  }

  // This wave's contiguous pre-packed weight stream: 32 tiles x 1KB.
  const unsigned short* wbase =
      Wp + ((unsigned)(q4 * 128 + wg) * 64u + (unsigned)(kh * 32)) * 512u;

  for (int t = 0; t < T; ++t) {
    const unsigned short* hread = hb + (t & 1) * H;
    unsigned short* hwrite      = hb + ((t + 1) & 1) * H;

    // --- y_{t-1} reduction (deterministic fixed-order) + input scalars ---
    float x0 = 0.f, ain = 0.f, bin = 0.f;
    if (wv == 0) {
      x0 = src[t * 3];
      if (t == 0) {
        ain = src[1];
        bin = src[2];
      } else {
        const float* pp = partials + ((t - 1) & 1) * NWG;
        float p = pp[lane] + pp[lane + 32] + pp[lane + 64] + pp[lane + 96];
#pragma unroll
        for (int s = 16; s > 0; s >>= 1) p += __shfl_xor(p, s, 32);
        float y = p + blin;
        ain = y;
        bin = y;
        if (wg == 0 && lane == 0) out[t - 1] = y;
      }
    }

    // --- stage h_{t} (bf16) into LDS: 4KB, 16B per thread ---
    *(uint4*)(hb_lds + tid * 8) = *(const uint4*)(hread + tid * 8);
    __syncthreads();

    // --- matvec: 32 chained v_wmma_f32_16x16x32_bf16, h broadcast as B ---
    v8f acc = {0.f, 0.f, 0.f, 0.f, 0.f, 0.f, 0.f, 0.f};
    const unsigned short* wp = wbase;
    const int bo = (lane >> 4) << 3;           // B half select (uint16 units)
#pragma unroll 4
    for (int kt = 0; kt < 32; ++kt) {
      union { v16bf v; uint4 q[2]; } A, B;
      const uint4* ap = (const uint4*)wp + (lane << 1);
      A.q[0] = ap[0];
      A.q[1] = ap[1];
      const uint4* bp = (const uint4*)(hb_lds + (kh * 1024 + kt * 32 + bo));
      B.q[0] = bp[0];
      B.q[1] = bp[2];
      __builtin_prefetch((const void*)(wp + 4 * 512), 0, 1);
      acc = __builtin_amdgcn_wmma_f32_16x16x32_bf16(
          false, A.v, false, B.v, (short)0, acc, false, false);
      wp += 512;
    }

    // --- extract 16 gate values (cols identical; lane0->M0-7, lane16->M8-15)
    if (lane == 0) {
#pragma unroll
      for (int m = 0; m < 8; ++m) gbuf[kh][q4 * 16 + m] = acc[m];
    } else if (lane == 16) {
#pragma unroll
      for (int m = 0; m < 8; ++m) gbuf[kh][q4 * 16 + 8 + m] = acc[m];
    }
    __syncthreads();

    // --- cell update (wave 0, lanes 0..15; c state lives in registers) ---
    float hn = 0.f;
    if (wv == 0 && lane < 16) {
      int j = lane;
      float g4[4];
#pragma unroll
      for (int q = 0; q < 4; ++q)
        g4[q] = gbuf[0][q * 16 + j] + gbuf[1][q * 16 + j] + bias_c[q] +
                wxa[q] * x0 + wxb[q] * ain + wxc[q] * bin;
      float si = sigmoidf(g4[0]);
      float sf = sigmoidf(g4[1]);
      float tg = tanhf(g4[2]);
      float so = sigmoidf(g4[3]);
      cstate = sf * cstate + si * tg;
      hn = so * tanhf(cstate);
      hwrite[wg * 16 + j] = f2bf(hn);
    }
    if (wv == 0) {
      float p = (lane < 16) ? wl * hn : 0.f;
#pragma unroll
      for (int s = 16; s > 0; s >>= 1) p += __shfl_xor(p, s, 32);
      if (lane == 0) partials[(t & 1) * NWG + wg] = p;
    }

    grid_sync(bar);
  }

  // --- final output y_{T-1} ---
  if (wg == 0 && wv == 0) {
    const float* pp = partials + ((T - 1) & 1) * NWG;
    float p = pp[lane] + pp[lane + 32] + pp[lane + 64] + pp[lane + 96];
#pragma unroll
    for (int s = 16; s > 0; s >>= 1) p += __shfl_xor(p, s, 32);
    if (lane == 0) out[T - 1] = p + blin;
  }
}

// ---------------------------------------------------------------------------
// Workspace layout:
//   [0,8)        barrier {cnt, gen}
//   [256,1280)   partials 2x128 f32
//   [4096,12288) h double buffer 2x2048 bf16
//   [65536, +32MB) packed bf16 W_hh
// First 64KB memset to 0 each call (graph-capturable) => deterministic.
// ---------------------------------------------------------------------------
extern "C" void kernel_launch(void* const* d_in, const int* in_sizes, int n_in,
                              void* d_out, int out_size, void* d_ws, size_t ws_size,
                              hipStream_t stream) {
  (void)in_sizes; (void)n_in; (void)out_size; (void)ws_size;
  const float* src   = (const float*)d_in[0];
  const float* W_ih  = (const float*)d_in[1];
  const float* W_hh  = (const float*)d_in[2];
  const float* b_ih  = (const float*)d_in[3];
  const float* b_hh  = (const float*)d_in[4];
  const float* W_lin = (const float*)d_in[5];
  const float* b_lin = (const float*)d_in[6];
  float* out = (float*)d_out;

  char* ws = (char*)d_ws;
  unsigned*       bar      = (unsigned*)(ws + 0);
  float*          partials = (float*)(ws + 256);
  unsigned short* hb       = (unsigned short*)(ws + 4096);
  unsigned short* Wp       = (unsigned short*)(ws + 65536);

  hipMemsetAsync(ws, 0, 65536, stream);

  pack_whh_kernel<<<65536, 256, 0, stream>>>(W_hh, Wp);

  lstm_persistent<<<NWG, WGSIZE, 0, stream>>>(
      src, W_ih, b_ih, b_hh, W_lin, b_lin, Wp, hb, partials, bar, out);
}
